// CausalSemigroupSelfAttentionSelective_65335042507234
// MI455X (gfx1250) — compile-verified, matrix operand order checked by reading, not run on hardware
//
#include <hip/hip_runtime.h>

#define T_SEQ 1024
#define C_DIM 768
#define H_NUM 12
#define D_HEAD 64
#define NT_NUM 8

typedef unsigned short u16;
typedef unsigned int u32;
typedef __attribute__((ext_vector_type(16))) __bf16 v16bf;
typedef __attribute__((ext_vector_type(8))) float v8f;
typedef __attribute__((ext_vector_type(4))) u32 u32x4;

struct HParams {
  float nhb;                 // -clip(beta,0.5,2.5)/2
  float inv4tau[NT_NUM];     // 1/(4*max(exp(log_tau),1e-6))
  float logw[NT_NUM];        // log(softmax(logit_w)+1e-12)
  float dp_coef;             // lam*dp_scale/(sqrt(D)*0.8)
  float lk_coef;             // (1-lam)*logk_scale
  float alpha;               // 0.1*sigmoid(dt_logit)
  float pad;
};

// ---------- bf16 helpers ----------
__device__ __forceinline__ u16 f2bf(float f) {
  u32 x = __builtin_bit_cast(u32, f);
  u32 r = x + 0x7FFFu + ((x >> 16) & 1u);   // round-to-nearest-even
  return (u16)(r >> 16);
}
__device__ __forceinline__ float bf2f(u16 u) {
  u32 x = ((u32)u) << 16;
  return __builtin_bit_cast(float, x);
}

// Fragment load matching the CDNA5 16-bit A/B operand layout:
// lane l -> row/col (l&15), half = l>>4; element e -> K = (e&7) + 8*half + 16*(e>>3)
// i.e. two contiguous 16B chunks at K = 8*half and K = 16+8*half.
__device__ __forceinline__ v16bf frag_ld(const u16* lds, int row, int pitch, int half) {
  union { v16bf v; u32x4 q[2]; } f;
  const u16* p = lds + row * pitch + half * 8;
  f.q[0] = *reinterpret_cast<const u32x4*>(p);
  f.q[1] = *reinterpret_cast<const u32x4*>(p + 16);
  return f.v;
}

__device__ __forceinline__ v8f wmma_bf16(v16bf a, v16bf b, v8f c) {
  return __builtin_amdgcn_wmma_f32_16x16x32_bf16(false, a, false, b, (short)0, c, false, false);
}

// ---------- per-head parameter setup ----------
__global__ void setup_params(const float* __restrict__ lam_p, const float* __restrict__ log_tau,
                             const float* __restrict__ logit_w, const float* __restrict__ beta,
                             const float* __restrict__ dp_scale, const float* __restrict__ logk_scale,
                             const float* __restrict__ dt_logit, HParams* __restrict__ par) {
  int h = threadIdx.x;
  if (h >= H_NUM) return;
  float lam = lam_p[0];
  float bc = fminf(fmaxf(beta[h], 0.5f), 2.5f);
  HParams p;
  p.nhb = -0.5f * bc;
  float m = -3.4e38f;
  for (int t = 0; t < NT_NUM; ++t) m = fmaxf(m, logit_w[h * NT_NUM + t]);
  float e[NT_NUM]; float se = 0.f;
  for (int t = 0; t < NT_NUM; ++t) { e[t] = expf(logit_w[h * NT_NUM + t] - m); se += e[t]; }
  for (int t = 0; t < NT_NUM; ++t) p.logw[t] = logf(e[t] / se + 1e-12f);
  for (int t = 0; t < NT_NUM; ++t)
    p.inv4tau[t] = 1.f / (4.f * fmaxf(expf(log_tau[h * NT_NUM + t]), 1e-6f));
  p.dp_coef = lam * dp_scale[h] / (8.f * 0.8f);   // sqrt(64)*DP_TEMP
  p.lk_coef = (1.f - lam) * logk_scale[h];
  p.alpha = 0.1f / (1.f + expf(-dt_logit[0]));    // ESR_ALPHA * sigmoid(dt_logit)
  p.pad = 0.f;
  par[h] = p;
}

// ---------- generic C = A @ B^T  (A:[M,K] f32, B:[N,K] f32, C:[M,N] f32), batched ----------
__global__ __launch_bounds__(256) void gemm_bt(
    const float* __restrict__ A, int lda, long long sA,
    const float* __restrict__ B, int ldb, long long sB,
    float* __restrict__ C, int ldc, long long sC, int K) {
  __shared__ alignas(16) u16 la[64 * 40];
  __shared__ alignas(16) u16 lb[64 * 40];
  int tid = threadIdx.x, lane = tid & 31, w = tid >> 5;
  int wr = w >> 1, wc = w & 1;           // 8 waves: 4 row-tiles x 2 col-halves -> 64x64 C tile
  const float* Ab = A + (long long)blockIdx.z * sA + (long long)blockIdx.y * 64 * lda;
  const float* Bb = B + (long long)blockIdx.z * sB + (long long)blockIdx.x * 64 * ldb;
  v8f acc0 = {0.f, 0.f, 0.f, 0.f, 0.f, 0.f, 0.f, 0.f};
  v8f acc1 = acc0;
  for (int kb = 0; kb < K; kb += 32) {
#pragma unroll
    for (int q = 0; q < 2; ++q) {
      int idx = tid + q * 256;
      int row = idx >> 3, kc = (idx & 7) * 4;
      float4 va = *reinterpret_cast<const float4*>(Ab + (long long)row * lda + kb + kc);
      u32 alo = (u32)f2bf(va.x) | ((u32)f2bf(va.y) << 16);
      u32 ahi = (u32)f2bf(va.z) | ((u32)f2bf(va.w) << 16);
      *reinterpret_cast<uint2*>(&la[row * 40 + kc]) = make_uint2(alo, ahi);
      float4 vb = *reinterpret_cast<const float4*>(Bb + (long long)row * ldb + kb + kc);
      u32 blo = (u32)f2bf(vb.x) | ((u32)f2bf(vb.y) << 16);
      u32 bhi = (u32)f2bf(vb.z) | ((u32)f2bf(vb.w) << 16);
      *reinterpret_cast<uint2*>(&lb[row * 40 + kc]) = make_uint2(blo, bhi);
    }
    __syncthreads();
    int hl = lane >> 4, ml = lane & 15;
    v16bf af = frag_ld(la, wr * 16 + ml, 40, hl);
    v16bf b0 = frag_ld(lb, wc * 32 + ml, 40, hl);
    v16bf b1 = frag_ld(lb, wc * 32 + 16 + ml, 40, hl);
    acc0 = wmma_bf16(af, b0, acc0);
    acc1 = wmma_bf16(af, b1, acc1);
    __syncthreads();
  }
  float* Cb = C + (long long)blockIdx.z * sC;
  int i0 = blockIdx.y * 64 + wr * 16 + (lane >> 4) * 8;
  int j0 = blockIdx.x * 64 + wc * 32 + (lane & 15);
#pragma unroll
  for (int r = 0; r < 8; ++r) {
    Cb[(long long)(i0 + r) * ldc + j0] = acc0[r];
    Cb[(long long)(i0 + r) * ldc + j0 + 16] = acc1[r];
  }
}

// ---------- head split, RMSNorm(g), bf16 pack, v^T (out_scale folded) ----------
__global__ __launch_bounds__(32) void split_heads(
    const float* __restrict__ qkvg, const float* __restrict__ g_norm_w,
    const float* __restrict__ out_scale,
    u16* __restrict__ qh, u16* __restrict__ kh, u16* __restrict__ gh,
    float* __restrict__ vt, float* __restrict__ g2) {
  int t = blockIdx.x, h = blockIdx.y, ln = threadIdx.x;   // wave32 block
  const float* base = qkvg + (size_t)t * (4 * C_DIM) + h * D_HEAD;
  float q0 = base[0 * C_DIM + ln], q1 = base[0 * C_DIM + ln + 32];
  float k0 = base[1 * C_DIM + ln], k1 = base[1 * C_DIM + ln + 32];
  float v0 = base[2 * C_DIM + ln], v1 = base[2 * C_DIM + ln + 32];
  float g0 = base[3 * C_DIM + ln], g1 = base[3 * C_DIM + ln + 32];
  float ss = g0 * g0 + g1 * g1;
#pragma unroll
  for (int off = 16; off > 0; off >>= 1) ss += __shfl_down(ss, off, 32);
  ss = __shfl(ss, 0, 32);
  float rms = rsqrtf(ss * (1.f / D_HEAD) + 1e-6f);
  float gn0 = g0 * rms * g_norm_w[ln];
  float gn1 = g1 * rms * g_norm_w[ln + 32];
  u16 gb0 = f2bf(gn0), gb1 = f2bf(gn1);
  float gq = bf2f(gb0) * bf2f(gb0) + bf2f(gb1) * bf2f(gb1);  // consistent with bf16 g.g^T
#pragma unroll
  for (int off = 16; off > 0; off >>= 1) gq += __shfl_down(gq, off, 32);
  size_t hd = (size_t)h * T_SEQ + t;
  if (ln == 0) g2[hd] = gq;
  size_t rowo = hd * D_HEAD;
  qh[rowo + ln] = f2bf(q0); qh[rowo + ln + 32] = f2bf(q1);
  kh[rowo + ln] = f2bf(k0); kh[rowo + ln + 32] = f2bf(k1);
  gh[rowo + ln] = gb0;      gh[rowo + ln + 32] = gb1;
  float os = out_scale[h];
  vt[((size_t)h * D_HEAD + ln) * T_SEQ + t] = v0 * os;
  vt[((size_t)h * D_HEAD + ln + 32) * T_SEQ + t] = v1 * os;
}

// ---------- fused scores: WMMA(q.k^T, g.g^T) + frac-log-kernel LSE + blend (lower tiles only) ----------
__global__ __launch_bounds__(256) void scores_kernel(
    const u16* __restrict__ qh, const u16* __restrict__ kh,
    const u16* __restrict__ gh, const float* __restrict__ g2,
    const HParams* __restrict__ par, float* __restrict__ P) {
  __shared__ alignas(16) u16 lq[64 * 72];
  __shared__ alignas(16) u16 lk[64 * 72];
  __shared__ alignas(16) u16 lgi[64 * 72];
  __shared__ alignas(16) u16 lgj[64 * 72];
  __shared__ float sg2i[64], sg2j[64];
  int tid = threadIdx.x, lane = tid & 31, w = tid >> 5;
  int wr = w >> 1, wc = w & 1;
  int h = blockIdx.y;
  int L = blockIdx.x;                                // lower-triangular tile index
  int ti = (int)((sqrtf(8.f * (float)L + 1.f) - 1.f) * 0.5f);
  while ((ti + 1) * (ti + 2) / 2 <= L) ++ti;
  while (ti * (ti + 1) / 2 > L) --ti;
  int tj = L - ti * (ti + 1) / 2;
  int m0 = ti * 64, n0 = tj * 64;
  const u16* gq = qh + ((size_t)h * T_SEQ + m0) * D_HEAD;
  const u16* gk = kh + ((size_t)h * T_SEQ + n0) * D_HEAD;
  const u16* gi = gh + ((size_t)h * T_SEQ + m0) * D_HEAD;
  const u16* gj = gh + ((size_t)h * T_SEQ + n0) * D_HEAD;
#pragma unroll
  for (int q = 0; q < 2; ++q) {
    int idx = tid + q * 256;
    int row = idx >> 3, c8 = (idx & 7) * 8;
    *reinterpret_cast<u32x4*>(&lq[row * 72 + c8])  = *reinterpret_cast<const u32x4*>(&gq[row * 64 + c8]);
    *reinterpret_cast<u32x4*>(&lk[row * 72 + c8])  = *reinterpret_cast<const u32x4*>(&gk[row * 64 + c8]);
    *reinterpret_cast<u32x4*>(&lgi[row * 72 + c8]) = *reinterpret_cast<const u32x4*>(&gi[row * 64 + c8]);
    *reinterpret_cast<u32x4*>(&lgj[row * 72 + c8]) = *reinterpret_cast<const u32x4*>(&gj[row * 64 + c8]);
  }
  if (tid < 64) sg2i[tid] = g2[(size_t)h * T_SEQ + m0 + tid];
  else if (tid < 128) sg2j[tid - 64] = g2[(size_t)h * T_SEQ + n0 + tid - 64];
  __syncthreads();
  v8f zero = {0.f, 0.f, 0.f, 0.f, 0.f, 0.f, 0.f, 0.f};
  v8f dp0 = zero, dp1 = zero, gg0 = zero, gg1 = zero;
  int hl = lane >> 4, ml = lane & 15;
#pragma unroll
  for (int kb = 0; kb < 64; kb += 32) {
    v16bf aq  = frag_ld(lq + kb, wr * 16 + ml, 72, hl);
    v16bf bk0 = frag_ld(lk + kb, wc * 32 + ml, 72, hl);
    v16bf bk1 = frag_ld(lk + kb, wc * 32 + 16 + ml, 72, hl);
    dp0 = wmma_bf16(aq, bk0, dp0);
    dp1 = wmma_bf16(aq, bk1, dp1);
    v16bf ag  = frag_ld(lgi + kb, wr * 16 + ml, 72, hl);
    v16bf bg0 = frag_ld(lgj + kb, wc * 32 + ml, 72, hl);
    v16bf bg1 = frag_ld(lgj + kb, wc * 32 + 16 + ml, 72, hl);
    gg0 = wmma_bf16(ag, bg0, gg0);
    gg1 = wmma_bf16(ag, bg1, gg1);
  }
  HParams hp = par[h];
  int ib = m0 + wr * 16 + hl * 8;
  int jb = n0 + wc * 32 + ml;
  float* Ph = P + (size_t)h * T_SEQ * T_SEQ;
#pragma unroll
  for (int r = 0; r < 8; ++r) {
    int i = ib + r;
#pragma unroll
    for (int seg = 0; seg < 2; ++seg) {
      int j = jb + seg * 16;
      if (j > i) continue;                       // strictly causal; upper never stored
      float dp = seg ? dp1[r] : dp0[r];
      float gg = seg ? gg1[r] : gg0[r];
      float s = fmaxf(sg2i[i - m0] + sg2j[j - n0] - 2.f * gg, 0.f);
      float av[NT_NUM]; float mv = -3.4e38f;
#pragma unroll
      for (int tq = 0; tq < NT_NUM; ++tq) {
        av[tq] = hp.nhb * log1pf(s * hp.inv4tau[tq]) + hp.logw[tq];
        mv = fmaxf(mv, av[tq]);
      }
      float se = 0.f;
#pragma unroll
      for (int tq = 0; tq < NT_NUM; ++tq) se += expf(av[tq] - mv);
      float logK = mv + logf(se);
      Ph[(size_t)i * T_SEQ + j] = hp.dp_coef * dp + hp.lk_coef * logK;
    }
  }
}

// ---------- fused causal softmax + reversibilization + alpha blend (in place) ----------
// Identity: P is exactly causal => masked P_adj is diagonal-only =>
// P_rev[i,j] = 0.5*(P_ij + d_ij*P_ii)/(0.5*sum_j P_ij + 0.5*P_ii);  pi (power iteration) is dead.
__device__ __forceinline__ float block_red(float v, float* sm, bool ismax) {
  int t = threadIdx.x;
  sm[t] = v; __syncthreads();
#pragma unroll
  for (int s = 128; s > 0; s >>= 1) {
    if (t < s) sm[t] = ismax ? fmaxf(sm[t], sm[t + s]) : (sm[t] + sm[t + s]);
    __syncthreads();
  }
  float r = sm[0]; __syncthreads();
  return r;
}

__global__ __launch_bounds__(256) void softmax_pfin(float* __restrict__ P,
                                                    const HParams* __restrict__ par) {
  __shared__ float sred[256];
  __shared__ float s_pii;
  int i = blockIdx.x, h = blockIdx.y, tid = threadIdx.x;
  float* row = P + ((size_t)h * T_SEQ + i) * T_SEQ;
  float v[4]; float lmax = -3.4e38f;
#pragma unroll
  for (int k = 0; k < 4; ++k) {
    int j = tid + k * 256;
    v[k] = (j <= i) ? row[j] : -3.4e38f;
    lmax = fmaxf(lmax, v[k]);
  }
  float mx = block_red(lmax, sred, true);
  float e[4]; float ls = 0.f;
#pragma unroll
  for (int k = 0; k < 4; ++k) {
    int j = tid + k * 256;
    e[k] = (j <= i) ? expf(fmaxf(v[k] - mx, -60.f)) : 0.f;
    ls += e[k];
  }
  float S = fmaxf(block_red(ls, sred, false), 1e-12f);
  float p[4]; float lp = 0.f;
#pragma unroll
  for (int k = 0; k < 4; ++k) {
    p[k] = e[k] / S;
    lp += p[k];
    int j = tid + k * 256;
    if (j == i) s_pii = p[k];
  }
  float srow = block_red(lp, sred, false);   // first barrier inside orders the s_pii store
  float pii = s_pii;
  float alpha = par[0].alpha;
  float rs = fmaxf(0.5f * srow + 0.5f * pii, 1e-12f);
#pragma unroll
  for (int k = 0; k < 4; ++k) {
    int j = tid + k * 256;
    float prev = 0.5f * (p[k] + ((j == i) ? pii : 0.f)) / rs;
    row[j] = (1.f - alpha) * p[k] + alpha * prev;   // zeros stay exact zeros for j>i
  }
}

extern "C" void kernel_launch(void* const* d_in, const int* in_sizes, int n_in,
                              void* d_out, int out_size, void* d_ws, size_t ws_size,
                              hipStream_t stream) {
  (void)in_sizes; (void)n_in; (void)out_size; (void)ws_size;
  const float* x         = (const float*)d_in[0];
  const float* W_qkvg    = (const float*)d_in[1];
  const float* W_out     = (const float*)d_in[2];
  const float* lam       = (const float*)d_in[3];
  const float* log_tau   = (const float*)d_in[4];
  const float* logit_w   = (const float*)d_in[5];
  const float* beta      = (const float*)d_in[6];
  const float* out_scale = (const float*)d_in[7];
  const float* dp_scale  = (const float*)d_in[8];
  const float* logk_scale= (const float*)d_in[9];
  const float* dt_logit  = (const float*)d_in[10];
  const float* g_norm_w  = (const float*)d_in[11];

  char* ws = (char*)d_ws;
  size_t off = 0;
  HParams* par = (HParams*)(ws + off); off += 4096;
  float* qkvg  = (float*)(ws + off);   off += (size_t)T_SEQ * 4 * C_DIM * 4;
  u16* qh      = (u16*)(ws + off);     off += (size_t)H_NUM * T_SEQ * D_HEAD * 2;
  u16* kh      = (u16*)(ws + off);     off += (size_t)H_NUM * T_SEQ * D_HEAD * 2;
  u16* gh      = (u16*)(ws + off);     off += (size_t)H_NUM * T_SEQ * D_HEAD * 2;
  float* vt    = (float*)(ws + off);   off += (size_t)H_NUM * D_HEAD * T_SEQ * 4;
  float* g2    = (float*)(ws + off);   off += (size_t)H_NUM * T_SEQ * 4;
  float* y     = (float*)(ws + off);   off += (size_t)T_SEQ * C_DIM * 4;
  float* P     = (float*)(ws + off);   off += (size_t)H_NUM * T_SEQ * T_SEQ * 4;

  setup_params<<<1, 16, 0, stream>>>(lam, log_tau, logit_w, beta, dp_scale, logk_scale,
                                     dt_logit, par);
  // qkvg = x @ W_qkvg^T
  gemm_bt<<<dim3(4 * C_DIM / 64, T_SEQ / 64, 1), 256, 0, stream>>>(
      x, C_DIM, 0, W_qkvg, C_DIM, 0, qkvg, 4 * C_DIM, 0, C_DIM);
  split_heads<<<dim3(T_SEQ, H_NUM), 32, 0, stream>>>(qkvg, g_norm_w, out_scale,
                                                     qh, kh, gh, vt, g2);
  // blended causal scores (lower-triangular tiles only): 136 tiles x 12 heads
  scores_kernel<<<dim3((T_SEQ / 64) * (T_SEQ / 64 + 1) / 2, H_NUM), 256, 0, stream>>>(
      qh, kh, gh, g2, par, P);
  softmax_pfin<<<dim3(T_SEQ, H_NUM), 256, 0, stream>>>(P, par);
  // y[:, h*64:(h+1)*64] = P_fin[h] @ v[h]   (out_scale folded into vt)
  gemm_bt<<<dim3(D_HEAD / 64, T_SEQ / 64, H_NUM), 256, 0, stream>>>(
      P, T_SEQ, (long long)T_SEQ * T_SEQ, vt, T_SEQ, (long long)D_HEAD * T_SEQ,
      y, C_DIM, D_HEAD, T_SEQ);
  // out = y @ W_out^T
  gemm_bt<<<dim3(C_DIM / 64, T_SEQ / 64, 1), 256, 0, stream>>>(
      y, C_DIM, 0, W_out, C_DIM, 0, (float*)d_out, C_DIM, 0, C_DIM);
}
